// Block_34522947125652
// MI455X (gfx1250) — compile-verified, matrix operand order checked by reading, not run on hardware
//
#include <hip/hip_runtime.h>
#include <hip/hip_bf16.h>
#include <math.h>

typedef __attribute__((ext_vector_type(16))) _Float16 v16h;
typedef __attribute__((ext_vector_type(8)))  _Float16 v8h;
typedef __attribute__((ext_vector_type(8)))  float    v8f;
typedef __attribute__((ext_vector_type(4)))  int      v4i;

#define NBATCH  4
#define CDIM    64
#define NWIN    1024      /* 4 * 16 * 16 windows */
#define SEQL    256       /* 16*16 tokens per window */
#define DI      128       /* D_INNER */
#define NTOK    262144    /* NWIN*SEQL == NBATCH*H*W pixels */

/* ---- workspace layout (bytes); ~512 MiB + 96KB, with phase reuse ---- */
#define O_XW  ((size_t)0)            /* f16 [NTOK,64]  ln2 out   ; later y3 f16 */
#define O_XM  ((size_t)33554432)     /* f32 [NTOK,128] xm/xmc    ; later hid f16 [NTOK,256] */
#define O_SZ  ((size_t)167772160)    /* f32 [NTOK,128] silu(z)   ; later attn f32 [NTOK,64] */
#define O_DT  ((size_t)301989888)    /* f32 [NTOK,128] dt then y ; later g f16 [NTOK,128] */
#define O_BM  ((size_t)436207616)    /* f32 [NTOK,16] */
#define O_CM  ((size_t)452984832)    /* f32 [NTOK,16] */
#define O_X2  ((size_t)469762048)    /* f32 [4,64,256,256] x+attn */
#define O_WH  ((size_t)536870912)    /* f16 weights: in(16384) out(8192) fin(16384) fout(8192) */

/* ---- CDNA5 async global->LDS (ASYNCcnt) feature probe ---- */
#if defined(__has_builtin)
#  if __has_builtin(__builtin_amdgcn_global_load_async_to_lds_b128) && \
      __has_builtin(__builtin_amdgcn_s_wait_asynccnt)
#    define HAVE_ASYNC_LDS 1
#  endif
#endif
#ifndef HAVE_ASYNC_LDS
#  define HAVE_ASYNC_LDS 0
#endif

/* builtin signature (from probe diagnostic): arg0 = AS1 (global) int4*, arg1 = LDS int4* */
typedef __attribute__((address_space(1))) v4i glb_v4i;
typedef __attribute__((address_space(3))) v4i lds_v4i;
__device__ __forceinline__ glb_v4i* to_glb4(const void* p){
  return (glb_v4i*)(size_t)p;
}
__device__ __forceinline__ lds_v4i* to_lds4(const void* p){
  /* generic LDS pointer: low 32 bits are the LDS offset (ISA 10.2 aperture rules) */
  return (lds_v4i*)(unsigned)(size_t)p;
}

__device__ __forceinline__ float sigmoidf_(float x){ return 1.f/(1.f+__expf(-x)); }
__device__ __forceinline__ float siluf_(float x){ return x*sigmoidf_(x); }
__device__ __forceinline__ float softplusf_(float x){ return (x>20.f)? x : log1pf(__expf(x)); }
__device__ __forceinline__ float geluf_(float x){ return 0.5f*x*(1.f+erff(x*0.70710678118f)); }

/* ---------- WMMA helpers (gfx1250, wave32, v_wmma_f32_16x16x32_f16) ----------
 * A tile 16x32 f16 per ISA 7.12.2: lanes 0-15 row M=lane hold K {0..7,16..23},
 * lanes 16-31 row M=lane-16 hold K {8..15,24..31}. Weights stored [N,K] row
 * major make B = W^T, with the identical per-lane K pattern over row N. */
__device__ __forceinline__ v16h ld_tile_h(const _Float16* row, int k0, int half){
  v8h lo = *(const v8h*)(row + k0 + half*8);
  v8h hi = *(const v8h*)(row + k0 + 16 + half*8);
  v16h r;
  #pragma unroll
  for (int i=0;i<8;++i){ r[i]=lo[i]; r[i+8]=hi[i]; }
  return r;
}
__device__ __forceinline__ v16h ld_tile_f(const float* row, int k0, int half){
  v8f lo = *(const v8f*)(row + k0 + half*8);
  v8f hi = *(const v8f*)(row + k0 + 16 + half*8);
  v16h r;
  #pragma unroll
  for (int i=0;i<8;++i){ r[i]=(_Float16)lo[i]; r[i+8]=(_Float16)hi[i]; }
  return r;
}
__device__ __forceinline__ v8f wmma_f16(v16h a, v16h b, v8f c){
  return __builtin_amdgcn_wmma_f32_16x16x32_f16(false, a, false, b, (short)0, c, false, false);
}

/* Per-wave GEMM strip: rows [M0,M0+16) x cols [N0,N0+16*NT), A row-major MxK,
 * W row-major NxK. Accumulate f32. */
template<int KTOT, int NT, bool AHALF, bool BHALF>
__device__ __forceinline__ void wave_gemm(const void* Aptr, int lda,
                                          const void* W, int ldw,
                                          int M0, int N0, int lane, v8f* acc)
{
  const int half = lane >> 4;
  const int r15  = lane & 15;
  v8f z = {};
  #pragma unroll
  for (int nt=0; nt<NT; ++nt) acc[nt] = z;
  #pragma unroll
  for (int k0=0; k0<KTOT; k0+=32){
    v16h a;
    if (AHALF) a = ld_tile_h((const _Float16*)Aptr + (size_t)(M0+r15)*lda, k0, half);
    else       a = ld_tile_f((const float*)Aptr    + (size_t)(M0+r15)*lda, k0, half);
    #pragma unroll
    for (int nt=0; nt<NT; ++nt){
      v16h b;
      if (BHALF) b = ld_tile_h((const _Float16*)W + (size_t)(N0 + nt*16 + r15)*ldw, k0, half);
      else       b = ld_tile_f((const float*)W    + (size_t)(N0 + nt*16 + r15)*ldw, k0, half);
      acc[nt] = wmma_f16(a, b, acc[nt]);
    }
  }
}

/* ---------------- K0: one-shot weight f32 -> f16 conversion ---------------- */
__global__ void k0_cvt_w(const float* __restrict__ w_in, const float* __restrict__ w_out,
                         const float* __restrict__ w_fin, const float* __restrict__ w_fout,
                         _Float16* __restrict__ dst)
{
  int i = blockIdx.x*256 + threadIdx.x;      /* 49152 elements total */
  if      (i < 16384) dst[i] = (_Float16)w_in[i];
  else if (i < 24576) dst[i] = (_Float16)w_out[i-16384];
  else if (i < 40960) dst[i] = (_Float16)w_fin[i-24576];
  else                dst[i] = (_Float16)w_fout[i-40960];
}

/* ---------------- K1: LN over channels + 16x16 window permute -> f16 ------- */
__global__ void k1_ln_window(const float* __restrict__ x, const float* __restrict__ lw,
                             const float* __restrict__ lb, _Float16* __restrict__ xw)
{
  int win = blockIdx.x, t = threadIdx.x;
  int b = win >> 8, wi = win & 255;
  int h = ((wi>>4)<<4) + (t>>4);
  int w = ((wi&15)<<4) + (t&15);
  size_t base = ((size_t)(b*64)) << 16;
  int sp = (h<<8) + w;
  float v[64]; float mu = 0.f;
  #pragma unroll
  for (int c=0;c<64;++c){ v[c] = x[base + ((size_t)c<<16) + sp]; mu += v[c]; }
  mu *= (1.f/64.f);
  float var = 0.f;
  #pragma unroll
  for (int c=0;c<64;++c){ float d = v[c]-mu; var += d*d; }
  var *= (1.f/64.f);
  float rs = rsqrtf(var + 1e-5f);
  _Float16* o = xw + ((size_t)win*SEQL + t)*64;
  #pragma unroll
  for (int c=0;c<64;++c) o[c] = (_Float16)((v[c]-mu)*rs*lw[c] + lb[c]);
}

/* ---------------- K2: in_proj GEMM [NTOK,64]x[64,256], split xm / silu(z) -- */
__global__ void k2_in_proj(const _Float16* __restrict__ xw, const _Float16* __restrict__ W,
                           float* __restrict__ xm, float* __restrict__ sz)
{
  int wave = threadIdx.x >> 5, lane = threadIdx.x & 31;
  int M0 = blockIdx.x*128 + wave*16;
  int N0 = blockIdx.y*64;
  if (M0 + 128 < NTOK) __builtin_prefetch(xw + (size_t)(M0+128)*64, 0, 1);
  v8f acc[4];
  wave_gemm<64,4,true,true>(xw, 64, W, 64, M0, N0, lane, acc);
  int half = lane>>4, col = lane&15;
  #pragma unroll
  for (int nt=0; nt<4; ++nt){
    int n = N0 + nt*16 + col;
    #pragma unroll
    for (int r=0;r<8;++r){
      int m = M0 + r + half*8;
      float v = acc[nt][r];
      if (n < 128) xm[(size_t)m*128 + n] = v;
      else         sz[(size_t)m*128 + (n-128)] = siluf_(v);
    }
  }
}

/* ---------------- K3: causal depthwise conv(k=4) + SiLU, in place ---------- */
__global__ void k3_conv(float* __restrict__ xm, const float* __restrict__ cw,
                        const float* __restrict__ cb)
{
  int win = blockIdx.x, d = threadIdx.x;
  float w0=cw[d*4+0], w1=cw[d*4+1], w2=cw[d*4+2], w3=cw[d*4+3], b=cb[d];
  float x1=0.f, x2=0.f, x3=0.f;   /* x[t-3], x[t-2], x[t-1] */
  float* p = xm + (size_t)win*SEQL*DI + d;
  for (int t=0;t<SEQL;++t){
    float xt = p[(size_t)t*DI];
    float v  = w3*xt + w2*x3 + w1*x2 + w0*x1 + b;
    p[(size_t)t*DI] = siluf_(v);
    x1=x2; x2=x3; x3=xt;
  }
}

/* ---------------- K4: x_proj (36 outs) + dt_proj + softplus, per token ----- */
__global__ void k4_xdt(const float* __restrict__ xmc, const float* __restrict__ xpw,
                       const float* __restrict__ dtw, const float* __restrict__ dtb,
                       float* __restrict__ dt, float* __restrict__ Bm, float* __restrict__ Cm)
{
  __shared__ float xs[128];
  __shared__ float dbl[36];
  int tok = blockIdx.x, d = threadIdx.x;
  xs[d] = xmc[(size_t)tok*128 + d];
  __syncthreads();
  if (d < 36){
    const float* wr = xpw + d*128;
    float s = 0.f;
    #pragma unroll 8
    for (int k=0;k<128;++k) s += xs[k]*wr[k];
    dbl[d] = s;
  }
  __syncthreads();
  float s = dtb[d];
  #pragma unroll
  for (int r=0;r<4;++r) s += dbl[r]*dtw[d*4+r];
  dt[(size_t)tok*128 + d] = softplusf_(s);
  if (d < 16)       Bm[(size_t)tok*16 + d]      = dbl[4+d];
  else if (d < 32)  Cm[(size_t)tok*16 + (d-16)] = dbl[20+(d-16)];
}

/* ---------------- K5: selective scan, h[16] in regs, B/C staged in LDS
 *                  via CDNA5 async global->LDS when available --------------- */
__global__ void k5_scan(float* __restrict__ dty, const float* __restrict__ xmc,
                        const float* __restrict__ sz, const float* __restrict__ Bm,
                        const float* __restrict__ Cm, const float* __restrict__ Alog,
                        const float* __restrict__ Dv)
{
  __shared__ __align__(16) float Bs[SEQL*16];
  __shared__ __align__(16) float Cs[SEQL*16];
  int win = blockIdx.x, d = threadIdx.x;
  const float* gB = Bm + (size_t)win*SEQL*16;
  const float* gC = Cm + (size_t)win*SEQL*16;
#if HAVE_ASYNC_LDS
  /* 128 lanes x b128 = 2KB/sweep; 8 sweeps per 16KB array. ASYNCcnt-tracked. */
  #pragma unroll
  for (int it=0; it<8; ++it){
    int i = it*512 + d*4;
    __builtin_amdgcn_global_load_async_to_lds_b128(to_glb4(gB + i), to_lds4(&Bs[i]), 0, 0);
    __builtin_amdgcn_global_load_async_to_lds_b128(to_glb4(gC + i), to_lds4(&Cs[i]), 0, 0);
  }
  __builtin_amdgcn_s_wait_asynccnt(0);
  __syncthreads();
#else
  for (int i=d; i<SEQL*16; i+=128){ Bs[i] = gB[i]; Cs[i] = gC[i]; }
  __syncthreads();
#endif
  float a[16], h[16];
  #pragma unroll
  for (int s=0;s<16;++s){ a[s] = -__expf(Alog[d*16+s]); h[s] = 0.f; }
  float dd = Dv[d];
  size_t base = (size_t)win*SEQL*DI + d;
  for (int t=0;t<SEQL;++t){
    size_t idx = base + (size_t)t*DI;
    float dtv = dty[idx];
    float u   = xmc[idx];
    float zv  = sz[idx];
    float du  = dtv*u, y = 0.f;
    #pragma unroll
    for (int s=0;s<16;++s){
      h[s] = h[s]*__expf(dtv*a[s]) + du*Bs[t*16+s];
      y   += h[s]*Cs[t*16+s];
    }
    dty[idx] = (y + u*dd)*zv;   /* in-place over dt: same slot, same step */
  }
}

/* ---------------- K6: out_proj GEMM [NTOK,128]x[128,64] -> token attn ------ */
__global__ void k6_out_proj(const float* __restrict__ y, const _Float16* __restrict__ W,
                            float* __restrict__ attn)
{
  int wave = threadIdx.x >> 5, lane = threadIdx.x & 31;
  int M0 = blockIdx.x*128 + wave*16;
  if (M0 + 128 < NTOK) __builtin_prefetch(y + (size_t)(M0+128)*128, 0, 1);
  v8f acc[4];
  wave_gemm<128,4,false,true>(y, 128, W, 128, M0, 0, lane, acc);
  int half = lane>>4, col = lane&15;
  #pragma unroll
  for (int nt=0; nt<4; ++nt){
    int n = nt*16 + col;
    #pragma unroll
    for (int r=0;r<8;++r){
      int m = M0 + r + half*8;
      attn[(size_t)m*64 + n] = acc[nt][r];
    }
  }
}

/* ---------------- K7: x2 = x + attn (un-window), LN3 -> f16 pixel-major ---- */
__global__ void k7_res_ln3(const float* __restrict__ x, const float* __restrict__ attn,
                           const float* __restrict__ lw, const float* __restrict__ lb,
                           float* __restrict__ x2, _Float16* __restrict__ y3)
{
  int p = blockIdx.x*256 + threadIdx.x;
  int b = p >> 16; int hw = p & 65535; int h = hw >> 8; int w = hw & 255;
  int win = (b<<8) + ((h>>4)<<4) + (w>>4);
  int t   = ((h&15)<<4) + (w&15);
  const float* ap = attn + ((size_t)win*SEQL + t)*64;
  size_t xbase = ((size_t)(b*64)) << 16;
  float v[64]; float mu = 0.f;
  #pragma unroll
  for (int c=0;c<64;++c){
    float s = x[xbase + ((size_t)c<<16) + hw] + ap[c];
    x2[xbase + ((size_t)c<<16) + hw] = s;
    v[c] = s; mu += s;
  }
  mu *= (1.f/64.f);
  float var = 0.f;
  #pragma unroll
  for (int c=0;c<64;++c){ float d = v[c]-mu; var += d*d; }
  var *= (1.f/64.f);
  float rs = rsqrtf(var + 1e-5f);
  _Float16* o = y3 + (size_t)p*64;
  #pragma unroll
  for (int c=0;c<64;++c) o[c] = (_Float16)((v[c]-mu)*rs*lw[c] + lb[c]);
}

/* ---------------- K8: ffn_in GEMM [NTOK,64]x[64,256] -> f16 hid ------------ */
__global__ void k8_ffn_in(const _Float16* __restrict__ y3, const _Float16* __restrict__ W,
                          _Float16* __restrict__ hid)
{
  int wave = threadIdx.x >> 5, lane = threadIdx.x & 31;
  int M0 = blockIdx.x*128 + wave*16;
  int N0 = blockIdx.y*64;
  if (M0 + 128 < NTOK) __builtin_prefetch(y3 + (size_t)(M0+128)*64, 0, 1);
  v8f acc[4];
  wave_gemm<64,4,true,true>(y3, 64, W, 64, M0, N0, lane, acc);
  int half = lane>>4, col = lane&15;
  #pragma unroll
  for (int nt=0; nt<4; ++nt){
    int n = N0 + nt*16 + col;
    #pragma unroll
    for (int r=0;r<8;++r){
      int m = M0 + r + half*8;
      hid[(size_t)m*256 + n] = (_Float16)acc[nt][r];
    }
  }
}

/* ---------------- K9: 3x3 depthwise (SAME) + exact-GELU gate --------------- */
__global__ void k9_dwconv(const _Float16* __restrict__ hid, const float* __restrict__ dw,
                          _Float16* __restrict__ g)
{
  int tid = threadIdx.x;
  int ch = tid & 127, which = tid >> 7;
  int p = blockIdx.x*2 + which;
  int b = p >> 16; int hw = p & 65535; int h = hw >> 8; int w = hw & 255;
  float a1 = 0.f, a2 = 0.f;
  const float* w1 = dw + ch*9;
  const float* w2 = dw + (ch+128)*9;
  #pragma unroll
  for (int dy=0; dy<3; ++dy){
    int hh = h + dy - 1; if ((unsigned)hh >= 256u) continue;
    #pragma unroll
    for (int dx=0; dx<3; ++dx){
      int ww = w + dx - 1; if ((unsigned)ww >= 256u) continue;
      size_t np = ((size_t)(b<<16) + (hh<<8) + ww) * 256;
      a1 += w1[dy*3+dx] * (float)hid[np + ch];
      a2 += w2[dy*3+dx] * (float)hid[np + ch + 128];
    }
  }
  g[(size_t)p*128 + ch] = (_Float16)(geluf_(a1)*a2);
}

/* ---------------- K10: ffn_out GEMM + final residual -> d_out -------------- */
__global__ void k10_ffn_out(const _Float16* __restrict__ g, const _Float16* __restrict__ W,
                            const float* __restrict__ x2, float* __restrict__ out)
{
  int wave = threadIdx.x >> 5, lane = threadIdx.x & 31;
  int M0 = blockIdx.x*128 + wave*16;
  if (M0 + 128 < NTOK) __builtin_prefetch(g + (size_t)(M0+128)*128, 0, 1);
  v8f acc[4];
  wave_gemm<128,4,true,true>(g, 128, W, 128, M0, 0, lane, acc);
  int half = lane>>4, col = lane&15;
  #pragma unroll
  for (int nt=0; nt<4; ++nt){
    int n = nt*16 + col;
    #pragma unroll
    for (int r=0;r<8;++r){
      int m  = M0 + r + half*8;           /* pixel index, raster order */
      int b  = m >> 16; int hw = m & 65535;
      size_t idx = ((size_t)(b*64 + n) << 16) + hw;
      out[idx] = x2[idx] + acc[nt][r];
    }
  }
}

extern "C" void kernel_launch(void* const* d_in, const int* in_sizes, int n_in,
                              void* d_out, int out_size, void* d_ws, size_t ws_size,
                              hipStream_t stream)
{
  const float* x       = (const float*)d_in[0];
  const float* ln2w    = (const float*)d_in[1];
  const float* ln2b    = (const float*)d_in[2];
  const float* ln3w    = (const float*)d_in[3];
  const float* ln3b    = (const float*)d_in[4];
  const float* inpw    = (const float*)d_in[5];
  const float* convw   = (const float*)d_in[6];
  const float* convb   = (const float*)d_in[7];
  const float* xprojw  = (const float*)d_in[8];
  const float* dtprojw = (const float*)d_in[9];
  const float* dtprojb = (const float*)d_in[10];
  const float* Alog    = (const float*)d_in[11];
  const float* Dp      = (const float*)d_in[12];
  const float* outpw   = (const float*)d_in[13];
  const float* ffninw  = (const float*)d_in[14];
  const float* ffndww  = (const float*)d_in[15];
  const float* ffnoutw = (const float*)d_in[16];

  char* ws = (char*)d_ws;
  _Float16* xw  = (_Float16*)(ws + O_XW);
  float*    xm  = (float*)(ws + O_XM);
  float*    sz  = (float*)(ws + O_SZ);
  float*    dty = (float*)(ws + O_DT);
  float*    Bm  = (float*)(ws + O_BM);
  float*    Cm  = (float*)(ws + O_CM);
  float*    x2  = (float*)(ws + O_X2);
  float*    attn = (float*)(ws + O_SZ);      /* reuse: sz dead after K5 */
  _Float16* y3   = (_Float16*)(ws + O_XW);   /* reuse: xw dead after K2 */
  _Float16* hid  = (_Float16*)(ws + O_XM);   /* reuse: xm dead after K5 */
  _Float16* g    = (_Float16*)(ws + O_DT);   /* reuse: y dead after K6  */
  _Float16* wh   = (_Float16*)(ws + O_WH);   /* f16 weights */
  _Float16* wh_in   = wh;
  _Float16* wh_out  = wh + 16384;
  _Float16* wh_fin  = wh + 24576;
  _Float16* wh_fout = wh + 40960;
  float*    out  = (float*)d_out;

  k0_cvt_w    <<<192, 256, 0, stream>>>(inpw, outpw, ffninw, ffnoutw, wh);
  k1_ln_window<<<NWIN, 256, 0, stream>>>(x, ln2w, ln2b, xw);
  k2_in_proj  <<<dim3(NTOK/128, 4), 256, 0, stream>>>(xw, wh_in, xm, sz);
  k3_conv     <<<NWIN, 128, 0, stream>>>(xm, convw, convb);
  k4_xdt      <<<NTOK, 128, 0, stream>>>(xm, xprojw, dtprojw, dtprojb, dty, Bm, Cm);
  k5_scan     <<<NWIN, 128, 0, stream>>>(dty, xm, sz, Bm, Cm, Alog, Dp);
  k6_out_proj <<<dim3(NTOK/128, 1), 256, 0, stream>>>(dty, wh_out, attn);
  k7_res_ln3  <<<NTOK/256, 256, 0, stream>>>(x, attn, ln3w, ln3b, x2, y3);
  k8_ffn_in   <<<dim3(NTOK/128, 4), 256, 0, stream>>>(y3, wh_fin, hid);
  k9_dwconv   <<<NTOK/2, 256, 0, stream>>>(hid, ffndww, g);
  k10_ffn_out <<<dim3(NTOK/128, 1), 256, 0, stream>>>(g, wh_fout, x2, out);
}
